// OptiHardTripletLoss_25598005084778
// MI455X (gfx1250) — compile-verified, matrix operand order checked by reading, not run on hardware
//
#include <hip/hip_runtime.h>
#include <hip/hip_bf16.h>

// ---------------------------------------------------------------------------
// OptiHardTripletLoss on MI455X (gfx1250, wave32, WMMA)
//
//   G = X Xt  (8192 x 8192 x 2048), bf16 WMMA w/ fp32 accumulation
//   D = sqrt(|x_i|^2 + |x_j|^2 - 2 G + eps)
//   pos_i = max same-person D, neg_i = min other-person D  (fused epilogue)
//   loss = mean softplus(margin + pos - neg)
//
// Design:
//  * Gram symmetry: only upper-triangle tiles (rt <= ct); each tile feeds BOTH
//    row-side and column-side mining (max/min idempotent -> no special casing).
//  * B tile staged with global_load_async_to_lds_b128 (ASYNCcnt) into a double
//    buffer; next tile's copy overlaps current tile's WMMAs; one barrier/step.
//  * Async source pointers advance by 64B/step -> precomputed once, bumped in
//    the loop (no per-iteration address rebuild).
//  * Explicit 2-deep rotating B fragment: ds_load of fragment c+1 issues
//    before WMMA c, keeping the DS pipe and matrix pipe overlapped
//    (s_wait_dscnt 0x1/0x2 instead of 0x0).
// ---------------------------------------------------------------------------

#define MARGIN_F 2.0f
#define EPS_F    1e-8f
#define DIMS     2048      // feature dim (reference setup)
#define TILE     128       // block tile (rows x cols)
#define KSTEP    32        // WMMA K depth per step (bf16)
#define LDS_PITCH 20       // uints per B-tile row (80B: 16B-aligned, 20-bank stride)
#define BUF_BYTES (TILE * LDS_PITCH * 4)   // 10240 B per buffer

typedef __attribute__((ext_vector_type(16))) __bf16 v16bf;
typedef __attribute__((ext_vector_type(8)))  float  v8f;

union Frag16 {
    v16bf v;
    uint4 q[2];
};

// ---------------------------------------------------------------------------
__global__ __launch_bounds__(256) void init_kernel(unsigned int* posb, unsigned int* negb,
                                                   float* out, int N) {
    int i = blockIdx.x * 256 + threadIdx.x;
    if (i < N) {
        posb[i] = 0u;           // 0.0f  (distances >= 0, int bit order == float order)
        negb[i] = 0x7F800000u;  // +inf
    }
    if (i == 0) out[0] = 0.0f;
}

// fp32 -> bf16 one-time convert (96 MiB of traffic; ~4 us at 23.3 TB/s)
__global__ __launch_bounds__(256) void cvt_kernel(const float* __restrict__ in,
                                                  __hip_bfloat16* __restrict__ out,
                                                  long total) {
    long idx    = (long)blockIdx.x * blockDim.x + threadIdx.x;
    long stride = (long)gridDim.x * blockDim.x;
    for (long i = idx; i < total; i += stride) out[i] = __float2bfloat16(in[i]);
}

// Row norms computed FROM the bf16 data so |x|^2 cancels consistently with the
// WMMA Gram diagonal.
__global__ __launch_bounds__(256) void norm_kernel(const __hip_bfloat16* __restrict__ f,
                                                   float* __restrict__ norms) {
    __shared__ float red[256];
    int row = blockIdx.x;
    const __hip_bfloat16* r = f + (long)row * DIMS;
    float s = 0.0f;
    for (int k = threadIdx.x; k < DIMS; k += 256) {
        float v = __bfloat162float(r[k]);
        s += v * v;
    }
    red[threadIdx.x] = s;
    __syncthreads();
    for (int m = 128; m > 0; m >>= 1) {
        if (threadIdx.x < m) red[threadIdx.x] += red[threadIdx.x + m];
        __syncthreads();
    }
    if (threadIdx.x == 0) norms[row] = red[0];
}

// two async 16B global->LDS copies (this thread's share of one 128x32 B tile)
__device__ __forceinline__ void stage2(const char* g0, const char* g1,
                                       unsigned l0, unsigned l1) {
    asm volatile("global_load_async_to_lds_b128 %0, %1, off"
                 :: "v"(l0), "v"(g0) : "memory");
    asm volatile("global_load_async_to_lds_b128 %0, %1, off"
                 :: "v"(l1), "v"(g1) : "memory");
}

// ---------------------------------------------------------------------------
// Main fused kernel.  Upper-triangle 128x128 Gram tiles; 8 waves/block, each
// wave = 16 rows x 128 cols = 8 v8f accumulators.
//
// Fragment packing (wave32, 16-bit A 16x32 / B 32x16, ISA 7.12.2):
//   lane L<16 : M/N = L,     K = {0..7} in q[0], {16..23} in q[1]
//   lane L>=16: M/N = L-16,  K = {8..15} in q[0], {24..31} in q[1]
// Loading rows of X identically into A and B computes X * Xt.
// C layout: vgpr r, lane L -> (M = r + 8*(L>=16), N = L&15).
// ---------------------------------------------------------------------------
__global__ __launch_bounds__(256) void gram_mine_kernel(const __hip_bfloat16* __restrict__ feat,
                                                        const float* __restrict__ norms,
                                                        unsigned int* __restrict__ posb,
                                                        unsigned int* __restrict__ negb,
                                                        const int* __restrict__ kimg_p) {
    if (blockIdx.y > blockIdx.x) return;          // symmetry: keep rt <= ct only

    __shared__ unsigned int ldsB[2 * TILE * LDS_PITCH];   // double buffer, 20 KB

    const int tid  = threadIdx.x;
    const int lane = tid & 31;
    const int wave = tid >> 5;
    const int half = (lane >= 16) ? 1 : 0;
    const int l16  = lane & 15;
    const int rowBase = blockIdx.y * TILE;        // rt
    const int colBase = blockIdx.x * TILE;        // ct  (rt <= ct)

    const int rowA = rowBase + wave * 16 + l16;   // this lane's A row
    const int koff = half ? 8 : 0;                // K sub-offset per lane half
    const __hip_bfloat16* arow = feat + (long)rowA * DIMS;

    // --- per-thread async-staging state: fixed LDS offsets, walking src ptrs
    const int r0 = tid >> 2;                      // tile row 0..63
    const int p0 = tid & 3;                       // 16B chunk within row
    const char* g0 = (const char*)(feat + (long)(colBase + r0) * DIMS + p0 * 8);
    const char* g1 = g0 + (long)64 * DIMS * 2;    // row r0+64, 64 rows below
    const unsigned lbase = (unsigned)(size_t)(&ldsB[0]);  // low 32b == LDS offset
    const unsigned o0 = lbase + (unsigned)(r0 * LDS_PITCH + p0 * 4) * 4u;
    const unsigned o1 = o0 + (unsigned)(64 * LDS_PITCH) * 4u;

    v8f acc[8];
    {
        v8f z = {};
        #pragma unroll
        for (int c = 0; c < 8; ++c) acc[c] = z;
    }

    const int steps = DIMS / KSTEP;               // 64
    stage2(g0, g1, o0, o1);                       // prologue prefetch -> buf0
    g0 += 64; g1 += 64;                           // KSTEP bf16 = 64 bytes

    for (int i = 0; i < steps; ++i) {
        const unsigned curOff = (i & 1) ? (unsigned)BUF_BYTES : 0u;
        const unsigned nxtOff = (i & 1) ? 0u : (unsigned)BUF_BYTES;
        unsigned int* cur = (unsigned int*)((char*)ldsB + curOff);

        // our async copies into `cur` complete; barrier makes everyone's
        // complete AND guarantees all waves finished reading `nxt` last round.
        asm volatile("s_wait_asynccnt 0x0" ::: "memory");
        __syncthreads();

        if (i + 1 < steps) {                      // overlaps this step's WMMAs
            stage2(g0, g1, o0 + nxtOff, o1 + nxtOff);
            g0 += 64; g1 += 64;
        }

        const int kb = i * KSTEP;

        // A fragment straight from global (rows unique per wave; L2-resident)
        Frag16 a;
        {
            const uint4* g = reinterpret_cast<const uint4*>(arow + kb + koff);
            a.q[0] = g[0];   // K = koff .. koff+7
            a.q[1] = g[2];   // K = 16+koff .. 16+koff+7
        }

        // rotating 2-deep B fragment: load c+1 before WMMA c so the DS pipe
        // stays ahead of the matrix pipe (dscnt 0x1/0x2 waits, not 0x0).
        const unsigned int* bbase = &cur[l16 * LDS_PITCH + (half ? 4 : 0)];
        Frag16 bf[2];
        bf[0].q[0] = *reinterpret_cast<const uint4*>(bbase);
        bf[0].q[1] = *reinterpret_cast<const uint4*>(bbase + 8);

        #pragma unroll
        for (int c = 0; c < 8; ++c) {
            if (c + 1 < 8) {
                const unsigned int* nb = bbase + (c + 1) * 16 * LDS_PITCH;
                bf[(c + 1) & 1].q[0] = *reinterpret_cast<const uint4*>(nb);
                bf[(c + 1) & 1].q[1] = *reinterpret_cast<const uint4*>(nb + 8);
            }
            acc[c] = __builtin_amdgcn_wmma_f32_16x16x32_bf16(
                false, a.v, false, bf[c & 1].v, (short)0, acc[c], false, false);
        }
    }

    // ---------------- epilogue: G -> distance -> hard mining ----------------
    const int kimg = kimg_p[0];                   // imgs per person (8)

    float normi[8];
    #pragma unroll
    for (int r = 0; r < 8; ++r)
        normi[r] = norms[rowBase + wave * 16 + half * 8 + r];

    float pmax[8], nmin[8];
    const float INF = __int_as_float(0x7F800000);
    #pragma unroll
    for (int r = 0; r < 8; ++r) { pmax[r] = 0.0f; nmin[r] = INF; }

    #pragma unroll
    for (int c = 0; c < 8; ++c) {
        int   j  = colBase + c * 16 + l16;
        float nj = norms[j];
        int   pj = j / kimg;

        float cpmax = 0.0f, cnmin = INF;          // column-side partials (16 rows)
        #pragma unroll
        for (int r = 0; r < 8; ++r) {
            int   i2 = rowBase + wave * 16 + half * 8 + r;
            float d2 = normi[r] + nj - 2.0f * acc[c][r];
            float dist = sqrtf(fmaxf(d2, 0.0f) + EPS_F);
            bool  same = (i2 / kimg) == pj;
            if (same) { pmax[r] = fmaxf(pmax[r], dist); cpmax = fmaxf(cpmax, dist); }
            else      { nmin[r] = fminf(nmin[r], dist); cnmin = fminf(cnmin, dist); }
        }
        // column-side: combine lane halves (M 0-7 with M 8-15), commit D[j][i]
        // candidates for row j.  Duplicates vs row-side on diagonal tiles are
        // harmless (idempotent max/min).
        cpmax = fmaxf(cpmax, __shfl_xor(cpmax, 16, 32));
        cnmin = fminf(cnmin, __shfl_xor(cnmin, 16, 32));
        if (half == 0) {
            atomicMax((int*)&posb[j], __float_as_int(cpmax));
            atomicMin((int*)&negb[j], __float_as_int(cnmin));
        }
    }

    // row-side: reduce across the 16-lane half holding that row's N values
    #pragma unroll
    for (int r = 0; r < 8; ++r) {
        #pragma unroll
        for (int m = 8; m >= 1; m >>= 1) {
            pmax[r] = fmaxf(pmax[r], __shfl_xor(pmax[r], m, 16));
            nmin[r] = fminf(nmin[r], __shfl_xor(nmin[r], m, 16));
        }
    }
    if (l16 == 0) {
        #pragma unroll
        for (int r = 0; r < 8; ++r) {
            int i2 = rowBase + wave * 16 + half * 8 + r;
            atomicMax((int*)&posb[i2], __float_as_int(pmax[r]));
            atomicMin((int*)&negb[i2], __float_as_int(nmin[r]));
        }
    }
}

// ---------------------------------------------------------------------------
__global__ __launch_bounds__(256) void loss_kernel(const unsigned int* __restrict__ posb,
                                                   const unsigned int* __restrict__ negb,
                                                   float* out, int N) {
    __shared__ float red[256];
    int idx = blockIdx.x * 256 + threadIdx.x;
    float s = 0.0f;
    for (int i = idx; i < N; i += gridDim.x * 256) {
        float p = __int_as_float((int)posb[i]);
        float n = __int_as_float((int)negb[i]);
        float x = MARGIN_F + p - n;
        s += fmaxf(x, 0.0f) + log1pf(expf(-fabsf(x)));   // stable softplus
    }
    red[threadIdx.x] = s;
    __syncthreads();
    for (int m = 128; m > 0; m >>= 1) {
        if (threadIdx.x < m) red[threadIdx.x] += red[threadIdx.x + m];
        __syncthreads();
    }
    if (threadIdx.x == 0) atomicAdd(out, red[0] / (float)N);
}

// ---------------------------------------------------------------------------
extern "C" void kernel_launch(void* const* d_in, const int* in_sizes, int n_in,
                              void* d_out, int out_size, void* d_ws, size_t ws_size,
                              hipStream_t stream) {
    const float* feat_f32 = (const float*)d_in[0];
    const int*   kimg     = (const int*)d_in[2];     // num_sample_imgs (device scalar)

    long total = (long)in_sizes[0];                  // N * DIMS
    int  N     = (int)(total / DIMS);                // 8192

    // workspace layout: bf16 features | norms | pos bits | neg bits
    char* ws = (char*)d_ws;
    __hip_bfloat16* featB = (__hip_bfloat16*)ws;
    size_t off = (size_t)N * DIMS * sizeof(__hip_bfloat16);
    float*        norms = (float*)(ws + off);        off += (size_t)N * sizeof(float);
    unsigned int* posb  = (unsigned int*)(ws + off); off += (size_t)N * sizeof(unsigned int);
    unsigned int* negb  = (unsigned int*)(ws + off);

    init_kernel<<<(N + 255) / 256, 256, 0, stream>>>(posb, negb, (float*)d_out, N);
    cvt_kernel <<<4096, 256, 0, stream>>>(feat_f32, featB, total);
    norm_kernel<<<N, 256, 0, stream>>>(featB, norms);

    dim3 grid(N / TILE, N / TILE);                   // 64x64; lower triangle exits early
    gram_mine_kernel<<<grid, 256, 0, stream>>>(featB, norms, posb, negb, kimg);

    loss_kernel<<<32, 256, 0, stream>>>(posb, negb, (float*)d_out, N);
}
